// GAT_83657372991832
// MI455X (gfx1250) — matrix operand order repaired and checked by reading, not compiled, not run on hardware
//
#include <hip/hip_runtime.h>
#include <math.h>

// ---------------------------------------------------------------------------
// Types for CDNA5 WMMA
// ---------------------------------------------------------------------------
typedef __attribute__((ext_vector_type(16))) __bf16 v16bf;
typedef __attribute__((ext_vector_type(16))) float  v16f;
typedef __attribute__((ext_vector_type(8)))  float  v8f;

__device__ __forceinline__ unsigned short f2bf(float f) {
  unsigned int u = __float_as_uint(f);
  u += 0x7fffu + ((u >> 16) & 1u);          // round-to-nearest-even
  return (unsigned short)(u >> 16);
}

__device__ __forceinline__ void atomicMaxF32(float* addr, float v) {
  // signed-int max for non-negative floats, unsigned-int min for negatives.
  if (v >= 0.0f) atomicMax((int*)addr, __float_as_int(v));
  else           atomicMin((unsigned int*)addr, __float_as_uint(v));
}

__device__ __forceinline__ float lrelu(float v) { return v > 0.0f ? v : 0.2f * v; }

// ---------------------------------------------------------------------------
// Weight prep: WT[c*K + k] = bf16(W[k*NC + c])  (transposed, bf16)
// ---------------------------------------------------------------------------
__global__ void wt_bf16_transpose(const float* __restrict__ W,
                                  unsigned short* __restrict__ WT, int K, int NC) {
  int t = blockIdx.x * blockDim.x + threadIdx.x;
  if (t >= K * NC) return;
  int k = t / NC, c = t - k * NC;
  WT[(size_t)c * K + k] = f2bf(W[t]);
}

// ---------------------------------------------------------------------------
// C[M,64] = A[M,K] @ W[K,64]; A fp32 row-major (converted on the fly),
// W pre-transposed bf16 (WT[64,K]).  One wave per 16-row stripe, producing
// all four 16x16 N-tiles: A fragment loaded once, reused by 4 WMMAs.
// Out-of-range rows read a zero-filled dummy row (no branches in data path).
// ---------------------------------------------------------------------------
__global__ __launch_bounds__(256) void gemm_bf16_wmma(
    const float* __restrict__ A, const unsigned short* __restrict__ WT,
    float* __restrict__ C, const float* __restrict__ zrow, int M, int K) {
  int wid  = (blockIdx.x * blockDim.x + threadIdx.x) >> 5;
  int lane = threadIdx.x & 31;
  int mtiles = (M + 15) >> 4;
  if (wid >= mtiles) return;

  int r    = lane & 15;
  int half = lane >> 4;
  int row  = wid * 16 + r;                    // A-fragment row for this lane
  bool rowok = row < M;
  const float* arow = rowok ? (A + (size_t)row * K) : zrow;
  // B columns for this lane, one per N-tile (pre-transposed bf16: contiguous K)
  const unsigned short* bc0 = WT + (size_t)(r)      * K;
  const unsigned short* bc1 = WT + (size_t)(16 + r) * K;
  const unsigned short* bc2 = WT + (size_t)(32 + r) * K;
  const unsigned short* bc3 = WT + (size_t)(48 + r) * K;

  v8f acc0 = {}, acc1 = {}, acc2 = {}, acc3 = {};
  for (int kb = 0; kb < K; kb += 32) {
    __builtin_prefetch((const void*)(arow + kb + 32), 0, 1);
    // ---- A fragment: 16x32 bf16 layout. Lane half 0: K={0..15},
    // half 1: K={16..31} -> one contiguous 16-float run.
    union { v16bf v; } ua;
    {
      const float4* p0 = (const float4*)(arow + kb + 16 * half);
      float4 a0 = p0[0], a1 = p0[1], a2 = p0[2], a3 = p0[3];
      v16f af;
      af[0]  = a0.x; af[1]  = a0.y; af[2]  = a0.z; af[3]  = a0.w;
      af[4]  = a1.x; af[5]  = a1.y; af[6]  = a1.z; af[7]  = a1.w;
      af[8]  = a2.x; af[9]  = a2.y; af[10] = a2.z; af[11] = a2.w;
      af[12] = a3.x; af[13] = a3.y; af[14] = a3.z; af[15] = a3.w;
      ua.v = __builtin_convertvector(af, v16bf);
    }
    // ---- B fragments: 32x16 bf16 layout: lane half selects K half, 16
    // contiguous bf16 = two aligned 16-byte loads per tile.
    int ko = kb + 16 * half;
    union { v16bf v; uint4 q[2]; } ub;
    ub.q[0] = ((const uint4*)(bc0 + ko))[0];
    ub.q[1] = ((const uint4*)(bc0 + ko))[1];
    acc0 = __builtin_amdgcn_wmma_f32_16x16x32_bf16(false, ua.v, false, ub.v,
                                                   (short)0, acc0, false, false);
    ub.q[0] = ((const uint4*)(bc1 + ko))[0];
    ub.q[1] = ((const uint4*)(bc1 + ko))[1];
    acc1 = __builtin_amdgcn_wmma_f32_16x16x32_bf16(false, ua.v, false, ub.v,
                                                   (short)0, acc1, false, false);
    ub.q[0] = ((const uint4*)(bc2 + ko))[0];
    ub.q[1] = ((const uint4*)(bc2 + ko))[1];
    acc2 = __builtin_amdgcn_wmma_f32_16x16x32_bf16(false, ua.v, false, ub.v,
                                                   (short)0, acc2, false, false);
    ub.q[0] = ((const uint4*)(bc3 + ko))[0];
    ub.q[1] = ((const uint4*)(bc3 + ko))[1];
    acc3 = __builtin_amdgcn_wmma_f32_16x16x32_bf16(false, ua.v, false, ub.v,
                                                   (short)0, acc3, false, false);
  }
  // C/D layout: lanes 0-15 -> M = 16*wid + j, lanes 16-31 -> +8; N = lane%16
#pragma unroll
  for (int j = 0; j < 8; ++j) {
    int m = wid * 16 + j + 8 * half;
    if (m < M) {
      float* crow = C + (size_t)m * 64 + r;
      crow[0]  = acc0[j];
      crow[16] = acc1[j];
      crow[32] = acc2[j];
      crow[48] = acc3[j];
    }
  }
}

// ---------------------------------------------------------------------------
// Utility fill
// ---------------------------------------------------------------------------
__global__ void fill_f32(float* __restrict__ p, float v, int n) {
  int i = blockIdx.x * blockDim.x + threadIdx.x;
  if (i < n) p[i] = v;
}

// ---------------------------------------------------------------------------
// Layer 1 (8 heads x 8 channels): per-node attention scores
// ---------------------------------------------------------------------------
__global__ void attn_scores_h8(const float* __restrict__ xl,
                               const float* __restrict__ att_src,
                               const float* __restrict__ att_dst,
                               float* __restrict__ as_, float* __restrict__ ad_,
                               int n) {
  int t = blockIdx.x * blockDim.x + threadIdx.x;
  if (t >= n * 8) return;
  int node = t >> 3, h = t & 7;
  const float* rowp = xl + (size_t)node * 64 + h * 8;
  const float* sp = att_src + h * 8;
  const float* dp = att_dst + h * 8;
  float vs = 0.0f, vd = 0.0f;
#pragma unroll
  for (int c = 0; c < 8; ++c) { float x = rowp[c]; vs += x * sp[c]; vd += x * dp[c]; }
  as_[t] = vs; ad_[t] = vd;
}

// edge enumeration helper: edges [0,E) from edge_index, [E, E+n) are self loops
__device__ __forceinline__ void edge_sd(const long long* __restrict__ ei, int E,
                                        int e, int& s, int& d) {
  if (e < E) { s = (int)ei[e]; d = (int)ei[(size_t)E + e]; }
  else       { s = d = e - E; }
}

__global__ void edge_max_h8(const long long* __restrict__ ei, int E, int n,
                            const float* __restrict__ as_, const float* __restrict__ ad_,
                            float* __restrict__ emax) {
  int t = blockIdx.x * blockDim.x + threadIdx.x;
  if (t >= (E + n) * 8) return;
  int e = t >> 3, h = t & 7, s, d;
  edge_sd(ei, E, e, s, d);
  float v = lrelu(as_[s * 8 + h] + ad_[d * 8 + h]);
  atomicMaxF32(&emax[d * 8 + h], v);
}

__global__ void edge_expsum_h8(const long long* __restrict__ ei, int E, int n,
                               const float* __restrict__ as_, const float* __restrict__ ad_,
                               const float* __restrict__ emax, float* __restrict__ denom) {
  int t = blockIdx.x * blockDim.x + threadIdx.x;
  if (t >= (E + n) * 8) return;
  int e = t >> 3, h = t & 7, s, d;
  edge_sd(ei, E, e, s, d);
  float v = lrelu(as_[s * 8 + h] + ad_[d * 8 + h]);
  float m = emax[d * 8 + h];
  if (!isfinite(m)) m = 0.0f;
  atomicAdd(&denom[d * 8 + h], __expf(v - m));
}

__global__ void edge_msg_h8(const long long* __restrict__ ei, int E, int n,
                            const float* __restrict__ as_, const float* __restrict__ ad_,
                            const float* __restrict__ emax, const float* __restrict__ denom,
                            const float* __restrict__ xl, float* __restrict__ out) {
  int t = blockIdx.x * blockDim.x + threadIdx.x;
  if (t >= (E + n) * 8) return;
  int e = t >> 3, h = t & 7, s, d;
  edge_sd(ei, E, e, s, d);
  float v = lrelu(as_[s * 8 + h] + ad_[d * 8 + h]);
  float m = emax[d * 8 + h];
  if (!isfinite(m)) m = 0.0f;
  float alpha = __expf(v - m) / fmaxf(denom[d * 8 + h], 1e-16f);
  const float4* xs = (const float4*)(xl + (size_t)s * 64 + h * 8);
  float4 x0 = xs[0], x1 = xs[1];
  float* od = out + (size_t)d * 64 + h * 8;
  atomicAdd(&od[0], x0.x * alpha);
  atomicAdd(&od[1], x0.y * alpha);
  atomicAdd(&od[2], x0.z * alpha);
  atomicAdd(&od[3], x0.w * alpha);
  atomicAdd(&od[4], x1.x * alpha);
  atomicAdd(&od[5], x1.y * alpha);
  atomicAdd(&od[6], x1.z * alpha);
  atomicAdd(&od[7], x1.w * alpha);
}

// h = ELU(out1 + b1), in place
__global__ void bias_elu(float* __restrict__ out1, const float* __restrict__ b1, int n64) {
  int i = blockIdx.x * blockDim.x + threadIdx.x;
  if (i >= n64) return;
  float v = out1[i] + b1[i & 63];
  out1[i] = v > 0.0f ? v : expm1f(v);
}

// ---------------------------------------------------------------------------
// Layer 2 (1 head x 64 channels)
// ---------------------------------------------------------------------------
__global__ void attn_scores_h1(const float* __restrict__ xl,
                               const float* __restrict__ att_src,
                               const float* __restrict__ att_dst,
                               float* __restrict__ as_, float* __restrict__ ad_, int n) {
  int node = blockIdx.x * blockDim.x + threadIdx.x;
  if (node >= n) return;
  const float* rowp = xl + (size_t)node * 64;
  float vs = 0.0f, vd = 0.0f;
#pragma unroll
  for (int c = 0; c < 64; ++c) { float x = rowp[c]; vs += x * att_src[c]; vd += x * att_dst[c]; }
  as_[node] = vs; ad_[node] = vd;
}

__global__ void edge_max_h1(const long long* __restrict__ ei, int E, int n,
                            const float* __restrict__ as_, const float* __restrict__ ad_,
                            float* __restrict__ emax) {
  int e = blockIdx.x * blockDim.x + threadIdx.x;
  if (e >= E + n) return;
  int s, d; edge_sd(ei, E, e, s, d);
  atomicMaxF32(&emax[d], lrelu(as_[s] + ad_[d]));
}

__global__ void edge_expsum_h1(const long long* __restrict__ ei, int E, int n,
                               const float* __restrict__ as_, const float* __restrict__ ad_,
                               const float* __restrict__ emax, float* __restrict__ denom) {
  int e = blockIdx.x * blockDim.x + threadIdx.x;
  if (e >= E + n) return;
  int s, d; edge_sd(ei, E, e, s, d);
  float m = emax[d];
  if (!isfinite(m)) m = 0.0f;
  atomicAdd(&denom[d], __expf(lrelu(as_[s] + ad_[d]) - m));
}

__global__ void edge_msg_h1(const long long* __restrict__ ei, int E, int n,
                            const float* __restrict__ as_, const float* __restrict__ ad_,
                            const float* __restrict__ emax, const float* __restrict__ denom,
                            const float* __restrict__ xl, float* __restrict__ out) {
  int t = blockIdx.x * blockDim.x + threadIdx.x;
  if (t >= (E + n) * 8) return;
  int e = t >> 3, chunk = t & 7, s, d;
  edge_sd(ei, E, e, s, d);
  float m = emax[d];
  if (!isfinite(m)) m = 0.0f;
  float alpha = __expf(lrelu(as_[s] + ad_[d]) - m) / fmaxf(denom[d], 1e-16f);
  const float4* xs = (const float4*)(xl + (size_t)s * 64 + chunk * 8);
  float4 x0 = xs[0], x1 = xs[1];
  float* od = out + (size_t)d * 64 + chunk * 8;
  atomicAdd(&od[0], x0.x * alpha);
  atomicAdd(&od[1], x0.y * alpha);
  atomicAdd(&od[2], x0.z * alpha);
  atomicAdd(&od[3], x0.w * alpha);
  atomicAdd(&od[4], x1.x * alpha);
  atomicAdd(&od[5], x1.y * alpha);
  atomicAdd(&od[6], x1.z * alpha);
  atomicAdd(&od[7], x1.w * alpha);
}

__global__ void final_logsoftmax(const float* __restrict__ out2,
                                 const float* __restrict__ b2,
                                 float* __restrict__ y, int n) {
  int node = blockIdx.x * blockDim.x + threadIdx.x;
  if (node >= n) return;
  const float* rowp = out2 + (size_t)node * 64;
  float v[64], mx = -INFINITY;
#pragma unroll
  for (int j = 0; j < 64; ++j) { v[j] = rowp[j] + b2[j]; mx = fmaxf(mx, v[j]); }
  float sum = 0.0f;
#pragma unroll
  for (int j = 0; j < 64; ++j) sum += __expf(v[j] - mx);
  float lse = mx + logf(sum);
  float* yp = y + (size_t)node * 64;
#pragma unroll
  for (int j = 0; j < 64; ++j) yp[j] = v[j] - lse;
}

// ---------------------------------------------------------------------------
// Launch
// ---------------------------------------------------------------------------
extern "C" void kernel_launch(void* const* d_in, const int* in_sizes, int n_in,
                              void* d_out, int out_size, void* d_ws, size_t ws_size,
                              hipStream_t stream) {
  const float*      x        = (const float*)d_in[0];
  const long long*  ei       = (const long long*)d_in[1];   // int64 [2,E]
  const float*      W1       = (const float*)d_in[2];
  const float*      att_src1 = (const float*)d_in[3];
  const float*      att_dst1 = (const float*)d_in[4];
  const float*      b1       = (const float*)d_in[5];
  const float*      W2       = (const float*)d_in[6];
  const float*      att_src2 = (const float*)d_in[7];
  const float*      att_dst2 = (const float*)d_in[8];
  const float*      b2       = (const float*)d_in[9];
  float*            y        = (float*)d_out;

  const int N  = in_sizes[0] / 512;
  const int E  = in_sizes[1] / 2;
  const int ET = E + N;

  // Workspace carving (256B aligned)
  char* cur = (char*)d_ws;
  auto alloc = [&](size_t bytes) -> void* {
    void* p = (void*)cur;
    cur += (bytes + 255) & ~(size_t)255;
    return p;
  };
  float* xl1    = (float*)alloc((size_t)N * 64 * 4);
  float* as1    = (float*)alloc((size_t)N * 8 * 4);
  float* ad1    = (float*)alloc((size_t)N * 8 * 4);
  float* emax1  = (float*)alloc((size_t)N * 8 * 4);
  float* denom1 = (float*)alloc((size_t)N * 8 * 4);
  float* out1   = (float*)alloc((size_t)N * 64 * 4);   // becomes h after bias+ELU
  float* xl2    = (float*)alloc((size_t)N * 64 * 4);
  float* as2    = (float*)alloc((size_t)N * 4);
  float* ad2    = (float*)alloc((size_t)N * 4);
  float* emax2  = (float*)alloc((size_t)N * 4);
  float* denom2 = (float*)alloc((size_t)N * 4);
  float* out2   = (float*)alloc((size_t)N * 64 * 4);
  unsigned short* WT1 = (unsigned short*)alloc((size_t)512 * 64 * 2);
  unsigned short* WT2 = (unsigned short*)alloc((size_t)64 * 64 * 2);
  float* zrow   = (float*)alloc((size_t)512 * 4);      // zero dummy row (max K)
  (void)ws_size; (void)n_in; (void)out_size;

  const int B256 = 256;
  auto blk = [](int t) { return (t + 255) / 256; };

  // ---- Prep: bf16-transposed weights + zero dummy row ----
  wt_bf16_transpose<<<blk(512 * 64), B256, 0, stream>>>(W1, WT1, 512, 64);
  wt_bf16_transpose<<<blk(64 * 64),  B256, 0, stream>>>(W2, WT2, 64, 64);
  fill_f32<<<2, B256, 0, stream>>>(zrow, 0.0f, 512);

  // ---- Layer 1 ----
  {
    int mtiles = (N + 15) / 16;
    gemm_bf16_wmma<<<(mtiles * 32 + 255) / 256, B256, 0, stream>>>(x, WT1, xl1, zrow, N, 512);
  }
  attn_scores_h8<<<blk(N * 8), B256, 0, stream>>>(xl1, att_src1, att_dst1, as1, ad1, N);
  fill_f32<<<blk(N * 8),  B256, 0, stream>>>(emax1, -INFINITY, N * 8);
  fill_f32<<<blk(N * 8),  B256, 0, stream>>>(denom1, 0.0f, N * 8);
  fill_f32<<<blk(N * 64), B256, 0, stream>>>(out1, 0.0f, N * 64);
  edge_max_h8   <<<blk(ET * 8), B256, 0, stream>>>(ei, E, N, as1, ad1, emax1);
  edge_expsum_h8<<<blk(ET * 8), B256, 0, stream>>>(ei, E, N, as1, ad1, emax1, denom1);
  edge_msg_h8   <<<blk(ET * 8), B256, 0, stream>>>(ei, E, N, as1, ad1, emax1, denom1, xl1, out1);
  bias_elu<<<blk(N * 64), B256, 0, stream>>>(out1, b1, N * 64);

  // ---- Layer 2 ----
  {
    int mtiles = (N + 15) / 16;
    gemm_bf16_wmma<<<(mtiles * 32 + 255) / 256, B256, 0, stream>>>(out1, WT2, xl2, zrow, N, 64);
  }
  attn_scores_h1<<<blk(N), B256, 0, stream>>>(xl2, att_src2, att_dst2, as2, ad2, N);
  fill_f32<<<blk(N),      B256, 0, stream>>>(emax2, -INFINITY, N);
  fill_f32<<<blk(N),      B256, 0, stream>>>(denom2, 0.0f, N);
  fill_f32<<<blk(N * 64), B256, 0, stream>>>(out2, 0.0f, N * 64);
  edge_max_h1   <<<blk(ET),     B256, 0, stream>>>(ei, E, N, as2, ad2, emax2);
  edge_expsum_h1<<<blk(ET),     B256, 0, stream>>>(ei, E, N, as2, ad2, emax2, denom2);
  edge_msg_h1   <<<blk(ET * 8), B256, 0, stream>>>(ei, E, N, as2, ad2, emax2, denom2, xl2, out2);

  final_logsoftmax<<<blk(N), B256, 0, stream>>>(out2, b2, y, N);
}
